// IlluminationGuidedAttention_54494545052382
// MI455X (gfx1250) — matrix operand order repaired and checked by reading, not compiled, not run on hardware
//
#include <hip/hip_runtime.h>
#include <hip/hip_bf16.h>

typedef __attribute__((ext_vector_type(16))) _Float16 v16h;
typedef __attribute__((ext_vector_type(8)))  _Float16 v8h;
typedef __attribute__((ext_vector_type(8)))  float    v8f;

#define BATCH 4
#define CH    64
#define HH    256
#define WW    256
#define HW    (HH*WW)

// ---------------------------------------------------------------------------
// Pack x: NCHW f32 -> NHWC f16 (LDS transpose, coalesced reads and writes)
// ---------------------------------------------------------------------------
__global__ __launch_bounds__(256) void pack_x_nhwc(const float* __restrict__ x,
                                                   _Float16* __restrict__ xh) {
    __shared__ _Float16 ls[64 * 65];
    int b  = blockIdx.y;
    int n0 = blockIdx.x * 64;
    int tid = threadIdx.x;
#pragma unroll
    for (int i = 0; i < 16; ++i) {
        int idx = i * 256 + tid;
        int c = idx >> 6, nn = idx & 63;
        ls[nn * 65 + c] = (_Float16)x[((size_t)(b * CH + c)) * HW + n0 + nn];
    }
    __syncthreads();
#pragma unroll
    for (int i = 0; i < 16; ++i) {
        int idx = i * 256 + tid;
        int nn = idx >> 6, c = idx & 63;
        xh[((size_t)(b * HW) + n0 + nn) * CH + c] = ls[nn * 65 + c];
    }
}

// ---------------------------------------------------------------------------
// Weight repacks (f32 -> f16, WMMA A-operand friendly layout)
// dense 3x3:  wp[tap][cib][co][kk]   (tap = dy*3+dx, kk = ci within 32-block)
// ---------------------------------------------------------------------------
__global__ void repack_w3(const float* __restrict__ w, _Float16* __restrict__ wp) {
    int n = blockIdx.x * 256 + threadIdx.x;
    if (n >= 9 * 2 * 64 * 32) return;
    int kk = n & 31, co = (n >> 5) & 63, cib = (n >> 11) & 1, tap = n >> 12;
    int ci = cib * 32 + kk;
    wp[n] = (_Float16)w[(co * 64 + ci) * 9 + tap];
}
__global__ void repack_w1(const float* __restrict__ w, _Float16* __restrict__ wp) {
    int n = blockIdx.x * 256 + threadIdx.x;
    if (n >= 2 * 64 * 32) return;
    int kk = n & 31, co = (n >> 5) & 63, cib = n >> 11;
    wp[n] = (_Float16)w[co * 64 + cib * 32 + kk];
}
__global__ void repack_dw(const float* __restrict__ w, float* __restrict__ wp) {
    int n = blockIdx.x * 256 + threadIdx.x;
    if (n >= 9 * 64) return;
    int c = n & 63, tap = n >> 6;
    wp[n] = w[c * 9 + tap];
}

// ---------------------------------------------------------------------------
// Dense 3x3 conv (implicit GEMM, WMMA f16->f32) + sigmoid, NHWC f16 in/out.
// WG tile: 64 co x 32 pos (8 waves of 16x16). K = 9 taps * 64 ci = 18 WMMAs.
// ---------------------------------------------------------------------------
__global__ __launch_bounds__(256) void conv3x3_sig_wmma(const _Float16* __restrict__ xh,
                                                        const _Float16* __restrict__ wp,
                                                        _Float16* __restrict__ out) {
    __shared__ __align__(32) _Float16 tile[3 * 34 * 64];   // halo rows x cols x ch
    __shared__ __align__(32) _Float16 ostage[32 * 64];
    int x0 = blockIdx.x * 32;
    int y  = blockIdx.y;
    int b  = blockIdx.z;
    int tid = threadIdx.x;

    for (int idx = tid; idx < 3 * 34 * 64; idx += 256) {
        int r   = idx / (34 * 64);
        int rem = idx - r * (34 * 64);
        int xl = rem >> 6, c = rem & 63;
        int yy = y + r - 1, xx = x0 + xl - 1;
        _Float16 v = (_Float16)0.0f;
        if (yy >= 0 && yy < HH && xx >= 0 && xx < WW)
            v = xh[((size_t)(b * HW) + yy * WW + xx) * CH + c];
        tile[idx] = v;
    }
    __syncthreads();

    int wave = tid >> 5, lane = tid & 31;
    int co0 = (wave & 3) * 16;
    int pn0 = (wave >> 2) * 16;
    int l15 = lane & 15, sel = lane >> 4;

    v8f acc = {};
#pragma unroll
    for (int tap = 0; tap < 9; ++tap) {
        int dy = tap / 3, dx = tap - dy * 3;
#pragma unroll
        for (int cib = 0; cib < 2; ++cib) {
            // A fragment: 16 co x 32 ci (contiguous k = sel*16 + s, matches B)
            const _Float16* ap = wp + (((tap * 2 + cib) * 64 + co0 + l15) << 5);
            v16h A = *(const v16h*)(ap + sel * 16);
            // B fragment: 32 ci x 16 pos, contiguous in LDS (NHWC-style tile)
            int pl = pn0 + l15;
            const _Float16* bp = &tile[(dy * 34 + pl + dx) * 64 + cib * 32 + sel * 16];
            v16h Bf = *(const v16h*)bp;
            acc = __builtin_amdgcn_wmma_f32_16x16x32_f16(false, A, false, Bf,
                                                         (short)0, acc, false, false);
        }
    }
    // sigmoid + stage D (VGPR r -> co0 + sel*8 + r, pos = pn0 + l15)
    int pl2 = pn0 + l15;
    int cob = co0 + sel * 8;
#pragma unroll
    for (int r = 0; r < 8; ++r) {
        float s = 1.0f / (1.0f + __expf(-acc[r]));
        ostage[pl2 * 64 + cob + r] = (_Float16)s;
    }
    __syncthreads();
    size_t base = ((size_t)(b * HW) + y * WW + x0) * CH;
    ((v8h*)(out + base))[tid] = ((const v8h*)ostage)[tid];
}

// ---------------------------------------------------------------------------
// Depthwise 3x3 conv + per-pixel channel mean. NHWC f16 in, NHWC f16 + f32 out
// ---------------------------------------------------------------------------
__global__ __launch_bounds__(256) void dwconv3x3_mean(const _Float16* __restrict__ in,
                                                      const float* __restrict__ wdw,
                                                      _Float16* __restrict__ outmap,
                                                      float* __restrict__ outscal) {
    int p = blockIdx.x * 256 + threadIdx.x;   // [0, B*HW)
    int b = p >> 16;
    int n = p & 65535;
    int y = n >> 8, x = n & 255;
    float acc[64];
#pragma unroll
    for (int c = 0; c < 64; ++c) acc[c] = 0.0f;
    for (int tap = 0; tap < 9; ++tap) {
        int yy = y + tap / 3 - 1, xx = x + tap % 3 - 1;
        if (yy < 0 || yy >= HH || xx < 0 || xx >= WW) continue;
        const _Float16* ip = in + ((size_t)(b * HW) + yy * WW + xx) * CH;
        const float* wpp = wdw + tap * 64;
#pragma unroll
        for (int c8 = 0; c8 < 8; ++c8) {
            v8h v = *(const v8h*)(ip + c8 * 8);
#pragma unroll
            for (int j = 0; j < 8; ++j)
                acc[c8 * 8 + j] += (float)v[j] * wpp[c8 * 8 + j];
        }
    }
    float s = 0.0f;
#pragma unroll
    for (int c = 0; c < 64; ++c) s += acc[c];
    outscal[p] = s * (1.0f / 64.0f);
    _Float16* op = outmap + (size_t)p * CH;
#pragma unroll
    for (int c8 = 0; c8 < 8; ++c8) {
        v8h v;
#pragma unroll
        for (int j = 0; j < 8; ++j) v[j] = (_Float16)acc[c8 * 8 + j];
        *(v8h*)(op + c8 * 8) = v;
    }
}

// ---------------------------------------------------------------------------
// 1x1 conv via WMMA, NHWC f16 -> NHWC f16 (safe in-place: WG pos-tile private)
// ---------------------------------------------------------------------------
__global__ __launch_bounds__(256) void conv1x1_wmma(const _Float16* __restrict__ in,
                                                    const _Float16* __restrict__ wp,
                                                    _Float16* __restrict__ out) {
    __shared__ __align__(32) _Float16 ostage[32 * 64];
    int bid = blockIdx.x;
    int b  = bid >> 11;            // 2048 tiles per image
    int p0 = (bid & 2047) * 32;
    int tid = threadIdx.x;
    int wave = tid >> 5, lane = tid & 31;
    int co0 = (wave & 3) * 16, pn0 = (wave >> 2) * 16;
    int l15 = lane & 15, sel = lane >> 4;

    v8f acc = {};
#pragma unroll
    for (int cib = 0; cib < 2; ++cib) {
        const _Float16* ap = wp + ((cib * 64 + co0 + l15) << 5);
        v16h A = *(const v16h*)(ap + sel * 16);
        const _Float16* bp = in + ((size_t)(b * HW) + p0 + pn0 + l15) * CH + cib * 32 + sel * 16;
        v16h Bf = *(const v16h*)bp;
        acc = __builtin_amdgcn_wmma_f32_16x16x32_f16(false, A, false, Bf,
                                                     (short)0, acc, false, false);
    }
    int pl2 = pn0 + l15;
    int cob = co0 + sel * 8;
#pragma unroll
    for (int r = 0; r < 8; ++r)
        ostage[pl2 * 64 + cob + r] = (_Float16)acc[r];
    __syncthreads();
    size_t base = ((size_t)(b * HW) + p0) * CH;
    ((v8h*)(out + base))[tid] = ((const v8h*)ostage)[tid];
}

// ---------------------------------------------------------------------------
// Projection 1x1 conv via WMMA: NHWC f16 in -> NCHW f32 out (coalesced rows)
// ---------------------------------------------------------------------------
__global__ __launch_bounds__(256) void conv1x1_proj_wmma(const _Float16* __restrict__ in,
                                                         const _Float16* __restrict__ wp,
                                                         float* __restrict__ out) {
    int bid = blockIdx.x;
    int b  = bid >> 11;
    int p0 = (bid & 2047) * 32;
    int tid = threadIdx.x;
    int wave = tid >> 5, lane = tid & 31;
    int co0 = (wave & 3) * 16, pn0 = (wave >> 2) * 16;
    int l15 = lane & 15, sel = lane >> 4;

    v8f acc = {};
#pragma unroll
    for (int cib = 0; cib < 2; ++cib) {
        const _Float16* ap = wp + ((cib * 64 + co0 + l15) << 5);
        v16h A = *(const v16h*)(ap + sel * 16);
        const _Float16* bp = in + ((size_t)(b * HW) + p0 + pn0 + l15) * CH + cib * 32 + sel * 16;
        v16h Bf = *(const v16h*)bp;
        acc = __builtin_amdgcn_wmma_f32_16x16x32_f16(false, A, false, Bf,
                                                     (short)0, acc, false, false);
    }
    int pos = p0 + pn0 + l15;
#pragma unroll
    for (int r = 0; r < 8; ++r) {
        int co = co0 + sel * 8 + r;
        out[((size_t)(b * CH + co)) * HW + pos] = acc[r];
    }
}

// ---------------------------------------------------------------------------
// Windowed attention: one WG per 8x8 window, one wave per head (wave32).
// l2-normalize q/k per (token, head), scale by illum/noise scalars, softmax.
// ---------------------------------------------------------------------------
__global__ __launch_bounds__(256) void win_attn(const _Float16* __restrict__ q,
                                                const _Float16* __restrict__ k,
                                                const _Float16* __restrict__ v,
                                                const float* __restrict__ iscal,
                                                const float* __restrict__ nscal,
                                                const float* __restrict__ temp,
                                                _Float16* __restrict__ out) {
    __shared__ float qf[64 * 64];
    __shared__ float kf[64 * 64];
    __shared__ float vf[64 * 64];
    __shared__ float isl[64], nsl[64];
    int x0 = blockIdx.x * 8, y0 = blockIdx.y * 8, b = blockIdx.z;
    int tid = threadIdx.x;

    if (tid < 64) {
        int n = (y0 + (tid >> 3)) * WW + x0 + (tid & 7);
        isl[tid] = iscal[b * HW + n];
        nsl[tid] = nscal[b * HW + n];
    }
    for (int idx = tid; idx < 4096; idx += 256) {
        int t = idx >> 6, c = idx & 63;
        int n = (y0 + (t >> 3)) * WW + x0 + (t & 7);
        size_t g = ((size_t)(b * HW) + n) * CH + c;
        qf[idx] = (float)q[g];
        kf[idx] = (float)k[g];
        vf[idx] = (float)v[g];
    }
    __syncthreads();
    for (int idx = tid; idx < 512; idx += 256) {   // (token, head) pairs
        int t = idx >> 3, h = idx & 7;
        float* qp = qf + t * 64 + h * 8;
        float ss = 0.0f;
#pragma unroll
        for (int j = 0; j < 8; ++j) ss += qp[j] * qp[j];
        float fq = (1.0f + isl[t]) / fmaxf(sqrtf(ss), 1e-12f);
#pragma unroll
        for (int j = 0; j < 8; ++j) qp[j] *= fq;
        float* kp = kf + t * 64 + h * 8;
        float ks = 0.0f;
#pragma unroll
        for (int j = 0; j < 8; ++j) ks += kp[j] * kp[j];
        float cf = fminf(fmaxf(1.0f - nsl[t], 0.0f), 1.0f);
        float fk = cf / fmaxf(sqrtf(ks), 1e-12f);
#pragma unroll
        for (int j = 0; j < 8; ++j) kp[j] *= fk;
    }
    __syncthreads();

    int h = tid >> 5;                 // wave == head
    int lane = tid & 31;
    float tmp = temp[h] * 0.35355339059327373f;   // * d^-0.5, d=8
    for (int q2 = 0; q2 < 2; ++q2) {
        int qi = lane + q2 * 32;
        float qreg[8];
#pragma unroll
        for (int j = 0; j < 8; ++j) qreg[j] = qf[qi * 64 + h * 8 + j];
        float m = -3.0e38f;
        for (int kj = 0; kj < 64; ++kj) {
            float dot = 0.0f;
#pragma unroll
            for (int j = 0; j < 8; ++j) dot += qreg[j] * kf[kj * 64 + h * 8 + j];
            m = fmaxf(m, dot * tmp);
        }
        float s = 0.0f, oacc[8];
#pragma unroll
        for (int j = 0; j < 8; ++j) oacc[j] = 0.0f;
        for (int kj = 0; kj < 64; ++kj) {
            float dot = 0.0f;
#pragma unroll
            for (int j = 0; j < 8; ++j) dot += qreg[j] * kf[kj * 64 + h * 8 + j];
            float e = __expf(dot * tmp - m);
            s += e;
#pragma unroll
            for (int j = 0; j < 8; ++j) oacc[j] += e * vf[kj * 64 + h * 8 + j];
        }
        float inv = 1.0f / s;
        int n = (y0 + (qi >> 3)) * WW + x0 + (qi & 7);
        _Float16* op = out + ((size_t)(b * HW) + n) * CH + h * 8;
        v8h ov;
#pragma unroll
        for (int j = 0; j < 8; ++j) ov[j] = (_Float16)(oacc[j] * inv);
        *(v8h*)op = ov;
    }
}

// ---------------------------------------------------------------------------
extern "C" void kernel_launch(void* const* d_in, const int* in_sizes, int n_in,
                              void* d_out, int out_size, void* d_ws, size_t ws_size,
                              hipStream_t stream) {
    (void)in_sizes; (void)n_in; (void)out_size; (void)ws_size;
    const float* x      = (const float*)d_in[0];
    const float* w_iem  = (const float*)d_in[1];
    const float* w_nem  = (const float*)d_in[2];
    const float* w_iemd = (const float*)d_in[3];
    const float* w_nemd = (const float*)d_in[4];
    const float* w_q    = (const float*)d_in[5];
    const float* w_k    = (const float*)d_in[6];
    const float* w_v    = (const float*)d_in[7];
    const float* w_proj = (const float*)d_in[8];
    const float* temp   = (const float*)d_in[9];
    float* out = (float*)d_out;

    char* ws = (char*)d_ws;
    size_t off = 0;
    auto take = [&](size_t bytes) -> char* {
        char* p = ws + off;
        off = (off + bytes + 255) & ~(size_t)255;
        return p;
    };
    const size_t mapH = sizeof(_Float16) * (size_t)BATCH * HW * CH;  // 33.5 MB
    _Float16* xh    = (_Float16*)take(mapH);   // x NHWC f16; later overwritten by v
    _Float16* sig   = (_Float16*)take(mapH);   // sigmoid(conv3) scratch; later attn out
    _Float16* illum = (_Float16*)take(mapH);   // illumination map; later q
    _Float16* noise = (_Float16*)take(mapH);   // noise map; later k
    float* iscal = (float*)take(sizeof(float) * (size_t)BATCH * HW);
    float* nscal = (float*)take(sizeof(float) * (size_t)BATCH * HW);
    _Float16* wp3i = (_Float16*)take(2 * 36864);
    _Float16* wp3n = (_Float16*)take(2 * 36864);
    _Float16* wp1q = (_Float16*)take(2 * 4096);
    _Float16* wp1k = (_Float16*)take(2 * 4096);
    _Float16* wp1v = (_Float16*)take(2 * 4096);
    _Float16* wp1p = (_Float16*)take(2 * 4096);
    float* wdwi = (float*)take(4 * 576);
    float* wdwn = (float*)take(4 * 576);

    pack_x_nhwc<<<dim3(HW / 64, BATCH), 256, 0, stream>>>(x, xh);
    repack_w3<<<144, 256, 0, stream>>>(w_iem, wp3i);
    repack_w3<<<144, 256, 0, stream>>>(w_nem, wp3n);
    repack_w1<<<16, 256, 0, stream>>>(w_q, wp1q);
    repack_w1<<<16, 256, 0, stream>>>(w_k, wp1k);
    repack_w1<<<16, 256, 0, stream>>>(w_v, wp1v);
    repack_w1<<<16, 256, 0, stream>>>(w_proj, wp1p);
    repack_dw<<<3, 256, 0, stream>>>(w_iemd, wdwi);
    repack_dw<<<3, 256, 0, stream>>>(w_nemd, wdwn);

    dim3 g3(WW / 32, HH, BATCH);
    int gdw = BATCH * HW / 256;
    int g1  = BATCH * HW / 32;

    conv3x3_sig_wmma<<<g3, 256, 0, stream>>>(xh, wp3i, sig);
    dwconv3x3_mean<<<gdw, 256, 0, stream>>>(sig, wdwi, illum, iscal);
    conv3x3_sig_wmma<<<g3, 256, 0, stream>>>(xh, wp3n, sig);
    dwconv3x3_mean<<<gdw, 256, 0, stream>>>(sig, wdwn, noise, nscal);

    conv1x1_wmma<<<g1, 256, 0, stream>>>(illum, wp1q, illum);   // q in place
    conv1x1_wmma<<<g1, 256, 0, stream>>>(noise, wp1k, noise);   // k in place
    conv1x1_wmma<<<g1, 256, 0, stream>>>(xh,    wp1v, xh);      // v in place

    win_attn<<<dim3(WW / 8, HH / 8, BATCH), 256, 0, stream>>>(illum, noise, xh,
                                                              iscal, nscal, temp, sig);
    conv1x1_proj_wmma<<<g1, 256, 0, stream>>>(sig, wp1p, out);
}
